// FEELVOS_26998164423251
// MI455X (gfx1250) — matrix operand order repaired and checked by reading, not compiled, not run on hardware
//
#include <hip/hip_runtime.h>

// ---------------------------------------------------------------------------
// FEELVOS pipeline for gfx1250 (MI455X).
// Heavy part: batched 2304x2304x(100->128) gram matrices through
// v_wmma_f32_16x16x32_f16 with squared norms folded into the K dimension,
// min-epilogue fused into the GEMM loop. Each wave is B-stationary over TWO
// q-tiles so every A fragment load feeds two WMMAs (2x arithmetic intensity
// vs round 1, and dual accumulator chains hide the WMMA RAW hazard).
// ---------------------------------------------------------------------------

typedef __attribute__((ext_vector_type(16))) _Float16 v16h;
typedef __attribute__((ext_vector_type(8)))  float    v8f;
typedef __attribute__((ext_vector_type(4)))  float    v4f;

#define HWDIM 48
#define NPIX  2304      // 48*48
#define EDIM  128       // 100 real dims + 2 folded-norm dims + 26 zero pad
#define NJ    4         // effective bc entries: {2,3,6,7}
#define NTILE 144       // 2304 / 16
#define QPW   2         // q-tiles per wave

// ---------------------------------------------------------------------------
// Generic direct 3x3 conv, pad=1, NCHW, optional ReLU
// ---------------------------------------------------------------------------
__global__ void conv3x3_k(const float* __restrict__ in, const float* __restrict__ wgt,
                          const float* __restrict__ bias, float* __restrict__ out,
                          int N, int Ci, int Co, int H, int W, int relu)
{
    int idx = blockIdx.x * blockDim.x + threadIdx.x;
    int total = N * Co * H * W;
    if (idx >= total) return;
    int x  = idx % W;
    int y  = (idx / W) % H;
    int co = (idx / (W * H)) % Co;
    int n  = idx / (W * H * Co);
    float acc = bias[co];
    for (int ci = 0; ci < Ci; ++ci) {
        const float* ip = in + ((size_t)(n * Ci + ci) * H) * W;
        const float* wp = wgt + ((size_t)(co * Ci + ci)) * 9;
        #pragma unroll
        for (int dh = -1; dh <= 1; ++dh) {
            int yy = y + dh;
            if (yy < 0 || yy >= H) continue;
            #pragma unroll
            for (int dw = -1; dw <= 1; ++dw) {
                int xx = x + dw;
                if (xx < 0 || xx >= W) continue;
                acc += ip[yy * W + xx] * wp[(dh + 1) * 3 + (dw + 1)];
            }
        }
    }
    if (relu) acc = fmaxf(acc, 0.0f);
    out[idx] = acc;
}

// 2x2 max pool, stride 2
__global__ void pool2x2_k(const float* __restrict__ in, float* __restrict__ out,
                          int N, int C, int Ho, int Wo)
{
    int idx = blockIdx.x * blockDim.x + threadIdx.x;
    int total = N * C * Ho * Wo;
    if (idx >= total) return;
    int wo = idx % Wo;
    int ho = (idx / Wo) % Ho;
    int nc = idx / (Wo * Ho);
    const float* ip = in + ((size_t)nc * 2 * Ho) * (2 * Wo);
    int Wi = 2 * Wo;
    float a = ip[(2 * ho) * Wi + 2 * wo];
    float b = ip[(2 * ho) * Wi + 2 * wo + 1];
    float c = ip[(2 * ho + 1) * Wi + 2 * wo];
    float d = ip[(2 * ho + 1) * Wi + 2 * wo + 1];
    out[idx] = fmaxf(fmaxf(a, b), fmaxf(c, d));
}

// conv3x3 over concat( upsample2x(A), S ), pad=1, ReLU. A is [N,Ca,H/2,W/2], S is [N,Cs,H,W].
__global__ void conv3x3_upcat_k(const float* __restrict__ A, const float* __restrict__ S,
                                const float* __restrict__ wgt, const float* __restrict__ bias,
                                float* __restrict__ out,
                                int N, int Ca, int Cs, int Co, int H, int W)
{
    int idx = blockIdx.x * blockDim.x + threadIdx.x;
    int total = N * Co * H * W;
    if (idx >= total) return;
    int x  = idx % W;
    int y  = (idx / W) % H;
    int co = (idx / (W * H)) % Co;
    int n  = idx / (W * H * Co);
    int H2 = H >> 1, W2 = W >> 1;
    int Cin = Ca + Cs;
    float acc = bias[co];
    for (int ci = 0; ci < Cin; ++ci) {
        const float* wp = wgt + ((size_t)(co * Cin + ci)) * 9;
        #pragma unroll
        for (int dh = -1; dh <= 1; ++dh) {
            int yy = y + dh;
            if (yy < 0 || yy >= H) continue;
            #pragma unroll
            for (int dw = -1; dw <= 1; ++dw) {
                int xx = x + dw;
                if (xx < 0 || xx >= W) continue;
                float v;
                if (ci < Ca)
                    v = A[(((size_t)(n * Ca + ci) * H2) + (yy >> 1)) * W2 + (xx >> 1)];
                else
                    v = S[(((size_t)(n * Cs + (ci - Ca)) * H) + yy) * W + xx];
                acc += v * wp[(dh + 1) * 3 + (dw + 1)];
            }
        }
    }
    out[idx] = fmaxf(acc, 0.0f);
}

// 1x1 conv, no relu
__global__ void conv1x1_k(const float* __restrict__ in, const float* __restrict__ wgt,
                          const float* __restrict__ bias, float* __restrict__ out,
                          int N, int Ci, int Co, int P)
{
    int idx = blockIdx.x * blockDim.x + threadIdx.x;
    int total = N * Co * P;
    if (idx >= total) return;
    int p  = idx % P;
    int co = (idx / P) % Co;
    int n  = idx / (P * Co);
    float acc = bias[co];
    for (int ci = 0; ci < Ci; ++ci)
        acc += in[((size_t)(n * Ci + ci)) * P + p] * wgt[co * Ci + ci];
    out[idx] = acc;
}

// ---------------------------------------------------------------------------
// Embedding conv (1 -> 100 ch, 3x3, pad 1) for bc in {2,3,6,7}, emitting f16
// rows [j][pixel][128] with squared norms folded into columns 100/101:
//   ref rows  : col100 = ||e||^2 , col101 = -0.5
//   cur rows  : col100 = -0.5   , col101 = ||e||^2
// so that  -2 * (a . b) = ||a||^2 + ||b||^2 - 2 G  (exact d^2 in the gram).
// ---------------------------------------------------------------------------
__global__ void emb_conv_k(const float* __restrict__ src,  // [8, 2304] planes
                           const float* __restrict__ ew,   // [100, 9]
                           const float* __restrict__ eb,   // [100]
                           _Float16* __restrict__ dst,     // [NJ, 2304, 128]
                           int mode)                       // 0 = ref, 1 = cur
{
    int idx = blockIdx.x * blockDim.x + threadIdx.x;
    if (idx >= NJ * NPIX) return;
    int j = idx / NPIX, p = idx % NPIX;
    int bc = 4 * (j >> 1) + 2 + (j & 1);
    const float* plane = src + (size_t)bc * NPIX;
    int h = p / HWDIM, w = p % HWDIM;

    float pix[9];
    #pragma unroll
    for (int dh = -1; dh <= 1; ++dh)
        #pragma unroll
        for (int dw = -1; dw <= 1; ++dw) {
            int hh = h + dh, ww = w + dw;
            pix[(dh + 1) * 3 + (dw + 1)] =
                (hh >= 0 && hh < HWDIM && ww >= 0 && ww < HWDIM) ? plane[hh * HWDIM + ww] : 0.0f;
        }

    _Float16* orow = dst + (size_t)idx * EDIM;
    float na = 0.0f;
    for (int e = 0; e < 100; ++e) {
        float acc = eb[e];
        #pragma unroll
        for (int k = 0; k < 9; ++k) acc += ew[e * 9 + k] * pix[k];
        na += acc * acc;
        orow[e] = (_Float16)acc;
    }
    if (mode == 0) { orow[100] = (_Float16)na;      orow[101] = (_Float16)(-0.5f); }
    else           { orow[100] = (_Float16)(-0.5f); orow[101] = (_Float16)na;      }
    #pragma unroll
    for (int e = 102; e < EDIM; ++e) orow[e] = (_Float16)0.0f;
}

// ---------------------------------------------------------------------------
// Global matching: one wave per (j, pair of q-tiles). Streams 144 P tiles;
// each A fragment feeds TWO independent WMMA accumulator chains (one per
// q-tile), then a max-epilogue -> dmin -> 1-2*sigmoid(-dmin).
// ---------------------------------------------------------------------------
__device__ __forceinline__ float match_map(float cmax)
{
    // d^2 = -2*cmax (norms folded into the gram); clamp, then 1 - 2*sigmoid(-d)
    float dmin = fmaxf(-2.0f * cmax, 0.0f);
    float e    = __expf(dmin);
    return 1.0f - 2.0f * __builtin_amdgcn_rcpf(1.0f + e);
}

__global__ __launch_bounds__(128)
void match_wmma_k(const _Float16* __restrict__ refE,  // [NJ, 2304, 128]
                  const _Float16* __restrict__ curE,  // [NJ, 2304, 128]
                  float* __restrict__ outv)           // [NJ, 2304]
{
    int lane = threadIdx.x & 31;
    int wave = blockIdx.x * 4 + (threadIdx.x >> 5);
    int j  = wave / (NTILE / QPW);
    int qp = wave % (NTILE / QPW);
    int qt0 = qp * QPW;                 // first of two adjacent q-tiles

    int lane16 = lane & 15;
    int hiA = (lane >> 4) * 8;    // A frag: upper half-wave holds K+8 group
    int hiB = (lane >> 4) * 8;    // B frag: upper half-wave holds K+8 group

    const _Float16* a = refE + (size_t)j * NPIX * EDIM;
    const _Float16* b = curE + (size_t)j * NPIX * EDIM;

    // B fragments: two q-tiles resident for the whole wave (B-stationary).
    v16h bf[QPW][4];
    #pragma unroll
    for (int qq = 0; qq < QPW; ++qq) {
        const _Float16* brow = b + (size_t)((qt0 + qq) * 16 + lane16) * EDIM + hiB;
        #pragma unroll
        for (int kk = 0; kk < 4; ++kk) {
            union { v16h v; v4f q[2]; } u;
            const _Float16* p = brow + kk * 32;
            u.q[0] = *(const v4f*)(p);
            u.q[1] = *(const v4f*)(p + 16);
            bf[qq][kk] = u.v;
        }
    }

    float cmax0 = -3.0e38f;
    float cmax1 = -3.0e38f;
    for (int pt = 0; pt < NTILE; ++pt) {
        const _Float16* arow = a + (size_t)(pt * 16 + lane16) * EDIM + hiA;
        v16h af[4];
        #pragma unroll
        for (int kk = 0; kk < 4; ++kk) {
            union { v16h v; v4f q[2]; } u;
            const _Float16* p = arow + kk * 32;
            u.q[0] = *(const v4f*)(p);
            u.q[1] = *(const v4f*)(p + 16);
            af[kk] = u.v;
        }
        v8f c0 = {};
        v8f c1 = {};
        #pragma unroll
        for (int kk = 0; kk < 4; ++kk) {
            c0 = __builtin_amdgcn_wmma_f32_16x16x32_f16(false, af[kk], false, bf[0][kk],
                                                        (short)0, c0, false, false);
            c1 = __builtin_amdgcn_wmma_f32_16x16x32_f16(false, af[kk], false, bf[1][kk],
                                                        (short)0, c1, false, false);
        }
        float t0 = fmaxf(fmaxf(fmaxf(c0[0], c0[1]), fmaxf(c0[2], c0[3])),
                         fmaxf(fmaxf(c0[4], c0[5]), fmaxf(c0[6], c0[7])));
        float t1 = fmaxf(fmaxf(fmaxf(c1[0], c1[1]), fmaxf(c1[2], c1[3])),
                         fmaxf(fmaxf(c1[4], c1[5]), fmaxf(c1[6], c1[7])));
        cmax0 = fmaxf(cmax0, t0);
        cmax1 = fmaxf(cmax1, t1);
    }
    // combine the two half-wave row groups (same q column, M and M+8 rows)
    cmax0 = fmaxf(cmax0, __shfl_xor(cmax0, 16, 32));
    cmax1 = fmaxf(cmax1, __shfl_xor(cmax1, 16, 32));
    if (lane < 16) {
        float* orow = outv + (size_t)j * NPIX + (size_t)qt0 * 16 + lane;
        orow[0]  = match_map(cmax0);
        orow[16] = match_map(cmax1);
    }
}

// ---------------------------------------------------------------------------
// Final dynamic-segmentation head: 3x3 conv over 7 gathered channels,
// only for the kept bc entries, writing the [2,2,48,48] output directly.
// ---------------------------------------------------------------------------
__global__ void dshead_k(const float* __restrict__ x3s,  // [8, 2304]
                         const float* __restrict__ x2,   // [8, 2304]
                         const float* __restrict__ gm,   // [NJ, 2304]
                         const float* __restrict__ lm,   // [NJ, 2304]
                         const float* __restrict__ dw,   // [7, 9]
                         const float* __restrict__ db,   // [1]
                         float* __restrict__ out)        // [2, 2, 2304]
{
    int idx = blockIdx.x * blockDim.x + threadIdx.x;
    if (idx >= NJ * NPIX) return;
    int j = idx / NPIX, p = idx % NPIX;
    int b = j >> 1, c = 2 + (j & 1);
    int h = p / HWDIM, w = p % HWDIM;
    float acc = db[0];
    #pragma unroll
    for (int dh = -1; dh <= 1; ++dh) {
        int hh = h + dh;
        if (hh < 0 || hh >= HWDIM) continue;
        #pragma unroll
        for (int dww = -1; dww <= 1; ++dww) {
            int ww = w + dww;
            if (ww < 0 || ww >= HWDIM) continue;
            int pp = hh * HWDIM + ww;
            int k  = (dh + 1) * 3 + (dww + 1);
            acc += dw[0 * 9 + k] * x3s[(size_t)(b * 4 + 0) * NPIX + pp];
            acc += dw[1 * 9 + k] * x3s[(size_t)(b * 4 + 1) * NPIX + pp];
            acc += dw[2 * 9 + k] * x3s[(size_t)(b * 4 + 2) * NPIX + pp];
            acc += dw[3 * 9 + k] * x3s[(size_t)(b * 4 + 3) * NPIX + pp];
            acc += dw[4 * 9 + k] * gm[(size_t)j * NPIX + pp];
            acc += dw[5 * 9 + k] * lm[(size_t)j * NPIX + pp];
            acc += dw[6 * 9 + k] * x2[(size_t)(b * 4 + c) * NPIX + pp];
        }
    }
    out[(size_t)(b * 2 + (j & 1)) * NPIX + p] = acc;
}

// ---------------------------------------------------------------------------
extern "C" void kernel_launch(void* const* d_in, const int* in_sizes, int n_in,
                              void* d_out, int out_size, void* d_ws, size_t ws_size,
                              hipStream_t stream)
{
    const float* x1     = (const float*)d_in[0];
    const float* x2     = (const float*)d_in[1];
    const float* x3     = (const float*)d_in[2];
    const float* enc1_w = (const float*)d_in[3];  const float* enc1_b = (const float*)d_in[4];
    const float* enc2_w = (const float*)d_in[5];  const float* enc2_b = (const float*)d_in[6];
    const float* bott_w = (const float*)d_in[7];  const float* bott_b = (const float*)d_in[8];
    const float* dec2_w = (const float*)d_in[9];  const float* dec2_b = (const float*)d_in[10];
    const float* dec1_w = (const float*)d_in[11]; const float* dec1_b = (const float*)d_in[12];
    const float* out_w  = (const float*)d_in[13]; const float* out_b  = (const float*)d_in[14];
    const float* emb_w  = (const float*)d_in[15]; const float* emb_b  = (const float*)d_in[16];
    const float* dsh_w  = (const float*)d_in[17]; const float* dsh_b  = (const float*)d_in[18];
    float* out = (float*)d_out;

    char*  ws  = (char*)d_ws;
    size_t off = 0;
    auto alloc = [&](size_t bytes) -> char* {
        char* p = ws + off;
        off += (bytes + 255) & ~(size_t)255;
        return p;
    };

    float* e1  = (float*)alloc((size_t)2 * 16 * 48 * 48 * 4);
    float* p1  = (float*)alloc((size_t)2 * 16 * 24 * 24 * 4);
    float* e2u = (float*)alloc((size_t)2 * 32 * 24 * 24 * 4);
    float* p2  = (float*)alloc((size_t)2 * 32 * 12 * 12 * 4);
    float* bt  = (float*)alloc((size_t)2 * 64 * 12 * 12 * 4);
    float* d2u = (float*)alloc((size_t)2 * 32 * 24 * 24 * 4);
    float* d1  = (float*)alloc((size_t)2 * 16 * 48 * 48 * 4);
    float* x3s = (float*)alloc((size_t)2 * 4 * NPIX * 4);
    _Float16* EA1 = (_Float16*)alloc((size_t)NJ * NPIX * EDIM * 2);
    _Float16* EA2 = (_Float16*)alloc((size_t)NJ * NPIX * EDIM * 2);
    _Float16* EB3 = (_Float16*)alloc((size_t)NJ * NPIX * EDIM * 2);
    float* gm = (float*)alloc((size_t)NJ * NPIX * 4);
    float* lm = (float*)alloc((size_t)NJ * NPIX * 4);
    (void)ws_size; (void)in_sizes; (void)n_in; (void)out_size;

    const int TB = 256;
    auto nb = [](int total, int tb) { return (total + tb - 1) / tb; };

    // ---- UNet on x3 ----
    conv3x3_k<<<nb(2 * 16 * 48 * 48, TB), TB, 0, stream>>>(x3, enc1_w, enc1_b, e1, 2, 3, 16, 48, 48, 1);
    pool2x2_k<<<nb(2 * 16 * 24 * 24, TB), TB, 0, stream>>>(e1, p1, 2, 16, 24, 24);
    conv3x3_k<<<nb(2 * 32 * 24 * 24, TB), TB, 0, stream>>>(p1, enc2_w, enc2_b, e2u, 2, 16, 32, 24, 24, 1);
    pool2x2_k<<<nb(2 * 32 * 12 * 12, TB), TB, 0, stream>>>(e2u, p2, 2, 32, 12, 12);
    conv3x3_k<<<nb(2 * 64 * 12 * 12, TB), TB, 0, stream>>>(p2, bott_w, bott_b, bt, 2, 32, 64, 12, 12, 1);
    conv3x3_upcat_k<<<nb(2 * 32 * 24 * 24, TB), TB, 0, stream>>>(bt, e2u, dec2_w, dec2_b, d2u, 2, 64, 32, 32, 24, 24);
    conv3x3_upcat_k<<<nb(2 * 16 * 48 * 48, TB), TB, 0, stream>>>(d2u, e1, dec1_w, dec1_b, d1, 2, 32, 16, 16, 48, 48);
    conv1x1_k<<<nb(2 * 4 * NPIX, TB), TB, 0, stream>>>(d1, out_w, out_b, x3s, 2, 16, 4, NPIX);

    // ---- Embeddings (only bc in {2,3,6,7}), f16 with folded norms ----
    emb_conv_k<<<nb(NJ * NPIX, TB), TB, 0, stream>>>(x1,  emb_w, emb_b, EA1, 0);
    emb_conv_k<<<nb(NJ * NPIX, TB), TB, 0, stream>>>(x2,  emb_w, emb_b, EA2, 0);
    emb_conv_k<<<nb(NJ * NPIX, TB), TB, 0, stream>>>(x3s, emb_w, emb_b, EB3, 1);

    // ---- Global matching via WMMA: 4 waves/block, 2 q-tiles per wave ----
    int mblocks = (NJ * (NTILE / QPW)) / 4;   // 288 waves / 4 = 72 blocks
    match_wmma_k<<<mblocks, 128, 0, stream>>>(EA1, EB3, gm);
    match_wmma_k<<<mblocks, 128, 0, stream>>>(EA2, EB3, lm);

    // ---- Head conv -> output [2,2,48,48] ----
    dshead_k<<<nb(NJ * NPIX, TB), TB, 0, stream>>>(x3s, x2, gm, lm, dsh_w, dsh_b, out);
}
